// BahdanauAttention_11416023072882
// MI455X (gfx1250) — compile-verified, matrix operand order checked by reading, not run on hardware
//
#include <hip/hip_runtime.h>
#include <hip/hip_bf16.h>
#include <math.h>

typedef __bf16 bf16_t;
typedef __attribute__((ext_vector_type(16))) __bf16 v16bf;
typedef __attribute__((ext_vector_type(8)))  __bf16 v8bf;
typedef __attribute__((ext_vector_type(4)))  __bf16 v4bf;
typedef __attribute__((ext_vector_type(8)))  float  v8f;

static constexpr int Dd = 2;
static constexpr int Bb = 16;
static constexpr int Ss = 2048;
static constexpr int DH = 1024;
static constexpr int EH = 1024;

// Hardware V_TANH_F32 (TRANS op) if the builtin is available on this toolchain.
#if defined(__has_builtin)
#if __has_builtin(__builtin_amdgcn_tanhf)
#define FAST_TANH(x) __builtin_amdgcn_tanhf(x)
#endif
#endif
#ifndef FAST_TANH
#define FAST_TANH(x) tanhf(x)
#endif

// ---------------------------------------------------------------------------
// f32 -> bf16 conversion, 4 elements / thread (grid-stride). n % 4 == 0.
// ---------------------------------------------------------------------------
__global__ void cvt_f32_bf16_kernel(const float* __restrict__ in,
                                    bf16_t* __restrict__ out, int n) {
    const float4* in4 = (const float4*)in;
    v4bf* out4 = (v4bf*)out;
    const int n4 = n >> 2;
    for (int i = blockIdx.x * blockDim.x + threadIdx.x; i < n4;
         i += gridDim.x * blockDim.x) {
        const float4 v = in4[i];
        v4bf o;
        o[0] = (bf16_t)v.x; o[1] = (bf16_t)v.y;
        o[2] = (bf16_t)v.z; o[3] = (bf16_t)v.w;
        out4[i] = o;
    }
}

// ---------------------------------------------------------------------------
// WMMA fragment loader: 16x32 bf16 tile slice for this lane.
// Lane holds K = {k0+sel .. k0+sel+7} and {k0+sel+16 .. k0+sel+23},
// sel = (lane>=16) ? 8 : 0, row/col = lane&15 (caller bakes into pointer).
// ---------------------------------------------------------------------------
__device__ __forceinline__ v16bf load_frag16(const bf16_t* p) {
    v8bf lo = *reinterpret_cast<const v8bf*>(p);
    v8bf hi = *reinterpret_cast<const v8bf*>(p + 16);
    v16bf r;
#pragma unroll
    for (int i = 0; i < 8; ++i) { r[i] = lo[i]; r[i + 8] = hi[i]; }
    return r;
}

// Computes c[t] += A(16xK) @ W[n_base+16t .. n_base+16t+15][:]^T for t=0..7.
// A: 16 rows, row-major, leading dim lda. W: NxK row-major (weights).
// All fragment loads for a k-step are issued before the 8 WMMAs so the
// compiler can clause the b128 loads and stagger the loadcnt waits.
__device__ __forceinline__ void wmma_strip(const bf16_t* __restrict__ A, int lda,
                                           const bf16_t* __restrict__ W, int ldw,
                                           int K, int n_base, v8f c[8]) {
    const int lane = threadIdx.x & 31;
    const int lo   = lane & 15;
    const int sel  = (lane & 16) ? 8 : 0;
    const bf16_t* arow = A + (size_t)lo * lda + sel;
    const bf16_t* wrow = W + (size_t)(n_base + lo) * ldw + sel;
    for (int k0 = 0; k0 < K; k0 += 32) {
        v16bf af = load_frag16(arow + k0);
        v16bf bf[8];
#pragma unroll
        for (int t = 0; t < 8; ++t)
            bf[t] = load_frag16(wrow + (size_t)t * 16 * ldw + k0);
#pragma unroll
        for (int t = 0; t < 8; ++t)
            c[t] = __builtin_amdgcn_wmma_f32_16x16x32_bf16(
                false, af, false, bf[t], (short)0, c[t], false, false);
    }
}

// ---------------------------------------------------------------------------
// Generic small GEMM: out[M x N] = A[M x K](bf16) @ W[N x K]^T(bf16) + bias
// grid: (N/1024, M/16), block 256 (8 waves, each wave = 128 columns)
// ---------------------------------------------------------------------------
__global__ __launch_bounds__(256) void gemm_bf16_kernel(
    const bf16_t* __restrict__ A, const bf16_t* __restrict__ W,
    const float* __restrict__ bias, float* __restrict__ out,
    int M, int N, int K) {
    const int m0   = blockIdx.y * 16;
    const int wave = threadIdx.x >> 5;
    const int lane = threadIdx.x & 31;
    const int lo   = lane & 15;
    const int n_base = blockIdx.x * 1024 + wave * 128;
    if (n_base >= N) return;

    v8f zero = {0.f, 0.f, 0.f, 0.f, 0.f, 0.f, 0.f, 0.f};
    v8f c[8];
#pragma unroll
    for (int t = 0; t < 8; ++t) c[t] = zero;

    wmma_strip(A + (size_t)m0 * K, K, W, K, K, n_base, c);

    const int rbase = (lane & 16) ? 8 : 0;
#pragma unroll
    for (int t = 0; t < 8; ++t) {
        const int n = n_base + t * 16 + lo;
        const float bv = bias[n];
#pragma unroll
        for (int j = 0; j < 8; ++j) {
            out[(size_t)(m0 + rbase + j) * N + n] = c[t][j] + bv;
        }
    }
}

// ---------------------------------------------------------------------------
// Fused: kh_tile = hs[b, s0:s0+16, :] @ k_w^T + k_b   (kept in registers)
//        e[d,b,s0+m] = sum_h tanh(qh[d,b,h] + kh[m,h]) * v_w[h] + v_b
// grid: (S/16, B), block 256 (8 waves x 128 cols = 1024 = DH)
// ---------------------------------------------------------------------------
__global__ __launch_bounds__(256) void kproj_score_kernel(
    const bf16_t* __restrict__ hs_bf, const bf16_t* __restrict__ kw_bf,
    const float* __restrict__ k_b, const float* __restrict__ qh,
    const float* __restrict__ v_w, const float* __restrict__ v_b,
    float* __restrict__ e_out) {
    const int s0   = blockIdx.x * 16;
    const int b    = blockIdx.y;
    const int wave = threadIdx.x >> 5;
    const int lane = threadIdx.x & 31;
    const int lo   = lane & 15;
    const int n_base = wave * 128;

    __shared__ float e_lds[2 * 16];
    if (threadIdx.x < 32) e_lds[threadIdx.x] = 0.f;
    __syncthreads();

    v8f zero = {0.f, 0.f, 0.f, 0.f, 0.f, 0.f, 0.f, 0.f};
    v8f c[8];
#pragma unroll
    for (int t = 0; t < 8; ++t) c[t] = zero;

    const bf16_t* A = hs_bf + ((size_t)b * Ss + s0) * EH;
    wmma_strip(A, EH, kw_bf, EH, EH, n_base, c);

    // score epilogue for d = 0 and d = 1
    float p0[8], p1[8];
#pragma unroll
    for (int j = 0; j < 8; ++j) { p0[j] = 0.f; p1[j] = 0.f; }

    const float* qh0 = qh + ((size_t)0 * Bb + b) * DH;
    const float* qh1 = qh + ((size_t)1 * Bb + b) * DH;
#pragma unroll
    for (int t = 0; t < 8; ++t) {
        const int h = n_base + t * 16 + lo;
        const float vwh = v_w[h];
        const float kbh = k_b[h];
        const float q0  = qh0[h];
        const float q1  = qh1[h];
#pragma unroll
        for (int j = 0; j < 8; ++j) {
            const float kv = c[t][j] + kbh;
            p0[j] += FAST_TANH(q0 + kv) * vwh;
            p1[j] += FAST_TANH(q1 + kv) * vwh;
        }
    }

    // reduce over the 16 columns held by each half-wave, then across waves
    const int rbase = (lane & 16) ? 8 : 0;
#pragma unroll
    for (int j = 0; j < 8; ++j) {
        float v0 = p0[j], v1 = p1[j];
#pragma unroll
        for (int off = 1; off < 16; off <<= 1) {
            v0 += __shfl_xor(v0, off, 32);
            v1 += __shfl_xor(v1, off, 32);
        }
        if (lo == 0) {
            atomicAdd(&e_lds[rbase + j], v0);
            atomicAdd(&e_lds[16 + rbase + j], v1);
        }
    }
    __syncthreads();

    if (threadIdx.x < 32) {
        const int d = threadIdx.x >> 4;
        const int m = threadIdx.x & 15;
        e_out[((size_t)d * Bb + b) * Ss + s0 + m] = e_lds[threadIdx.x] + v_b[0];
    }
}

// ---------------------------------------------------------------------------
// Row softmax over S: at = softmax(e).  grid: D*B rows, block 256.
// ---------------------------------------------------------------------------
__global__ __launch_bounds__(256) void softmax_kernel(
    const float* __restrict__ e, float* __restrict__ at) {
    const int row = blockIdx.x;
    const float* er = e + (size_t)row * Ss;
    float* ar = at + (size_t)row * Ss;
    __shared__ float red[256];

    float m = -INFINITY;
    for (int s = threadIdx.x; s < Ss; s += 256) m = fmaxf(m, er[s]);
    red[threadIdx.x] = m;
    __syncthreads();
    for (int off = 128; off > 0; off >>= 1) {
        if (threadIdx.x < off)
            red[threadIdx.x] = fmaxf(red[threadIdx.x], red[threadIdx.x + off]);
        __syncthreads();
    }
    m = red[0];
    __syncthreads();

    float sum = 0.f;
    for (int s = threadIdx.x; s < Ss; s += 256) {
        const float v = __expf(er[s] - m);
        ar[s] = v;
        sum += v;
    }
    red[threadIdx.x] = sum;
    __syncthreads();
    for (int off = 128; off > 0; off >>= 1) {
        if (threadIdx.x < off) red[threadIdx.x] += red[threadIdx.x + off];
        __syncthreads();
    }
    const float inv = 1.f / red[0];
    for (int s = threadIdx.x; s < Ss; s += 256) ar[s] *= inv;
}

// ---------------------------------------------------------------------------
// ctx[d,b,g] = sum_s at[d,b,s] * hs[b,s,g]  (read hs once for both d)
// grid: (EH/256, B), block 256.  Output stored as bf16 (feeds final GEMM).
// ---------------------------------------------------------------------------
__global__ __launch_bounds__(256) void ctx_kernel(
    const float* __restrict__ at, const float* __restrict__ hs,
    bf16_t* __restrict__ ctx_bf) {
    const int b = blockIdx.y;
    const int g = blockIdx.x * 256 + threadIdx.x;
    const float* a0 = at + ((size_t)0 * Bb + b) * Ss;
    const float* a1 = at + ((size_t)1 * Bb + b) * Ss;
    const float* h  = hs + (size_t)b * Ss * EH + g;
    float acc0 = 0.f, acc1 = 0.f;
    for (int s = 0; s < Ss; ++s) {
        const float hv = h[(size_t)s * EH];
        acc0 += a0[s] * hv;
        acc1 += a1[s] * hv;
    }
    ctx_bf[((size_t)0 * Bb + b) * EH + g] = (bf16_t)acc0;
    ctx_bf[((size_t)1 * Bb + b) * EH + g] = (bf16_t)acc1;
}

// ---------------------------------------------------------------------------
// launcher
// ---------------------------------------------------------------------------
extern "C" void kernel_launch(void* const* d_in, const int* in_sizes, int n_in,
                              void* d_out, int out_size, void* d_ws, size_t ws_size,
                              hipStream_t stream) {
    const float* hs  = (const float*)d_in[0];  // (B,S,EH)
    const float* ht  = (const float*)d_in[1];  // (D,B,DH)
    const float* q_w = (const float*)d_in[2];  // (DH,DH)
    const float* q_b = (const float*)d_in[3];  // (DH,)
    const float* k_w = (const float*)d_in[4];  // (DH,EH)
    const float* k_b = (const float*)d_in[5];  // (DH,)
    const float* v_w = (const float*)d_in[6];  // (1,DH)
    const float* v_b = (const float*)d_in[7];  // (1,)
    const float* c_w = (const float*)d_in[8];  // (DH,EH)
    const float* c_b = (const float*)d_in[9];  // (DH,)

    float* out_c  = (float*)d_out;                 // (D,B,DH) = 32768
    float* out_at = out_c + (size_t)Dd * Bb * DH;  // (D,B,S)  = 65536

    char* ws = (char*)d_ws;
    bf16_t* hs_bf  = (bf16_t*)(ws + 0);           // 67108864 B
    bf16_t* kw_bf  = (bf16_t*)(ws + 67108864);    //  2097152 B
    bf16_t* qw_bf  = (bf16_t*)(ws + 69206016);    //  2097152 B
    bf16_t* cw_bf  = (bf16_t*)(ws + 71303168);    //  2097152 B
    bf16_t* ht_bf  = (bf16_t*)(ws + 73400320);    //    65536 B
    bf16_t* ctx_bf = (bf16_t*)(ws + 73465856);    //    65536 B
    float*  qh     = (float*) (ws + 73531392);    //   131072 B
    float*  e      = (float*) (ws + 73662464);    //   262144 B

    // 1) f32 -> bf16 conversions
    cvt_f32_bf16_kernel<<<8192, 256, 0, stream>>>(hs, hs_bf, Bb * Ss * EH);
    cvt_f32_bf16_kernel<<<512, 256, 0, stream>>>(k_w, kw_bf, DH * EH);
    cvt_f32_bf16_kernel<<<512, 256, 0, stream>>>(q_w, qw_bf, DH * DH);
    cvt_f32_bf16_kernel<<<512, 256, 0, stream>>>(c_w, cw_bf, DH * EH);
    cvt_f32_bf16_kernel<<<32, 256, 0, stream>>>(ht, ht_bf, Dd * Bb * DH);

    // 2) qh = ht @ q_w^T + q_b   (M=32, N=DH, K=DH)
    gemm_bf16_kernel<<<dim3(1, 2), 256, 0, stream>>>(
        ht_bf, qw_bf, q_b, qh, Dd * Bb, DH, DH);

    // 3) fused k-projection + tanh score -> e (D,B,S)
    kproj_score_kernel<<<dim3(Ss / 16, Bb), 256, 0, stream>>>(
        hs_bf, kw_bf, k_b, qh, v_w, v_b, e);

    // 4) softmax over S -> at output
    softmax_kernel<<<Dd * Bb, 256, 0, stream>>>(e, out_at);

    // 5) ctx = at @ hs  (bf16 output)
    ctx_kernel<<<dim3(EH / 256, Bb), 256, 0, stream>>>(out_at, hs, ctx_bf);

    // 6) c = ctx @ c_w^T + c_b   (M=32, N=DH, K=EH)
    gemm_bf16_kernel<<<dim3(1, 2), 256, 0, stream>>>(
        ctx_bf, cw_bf, c_b, out_c, Dd * Bb, DH, EH);
}
